// MaskedMultiHeadAttention_5652176961918
// MI455X (gfx1250) — matrix operand order repaired and checked by reading, not compiled, not run on hardware
//
#include <hip/hip_runtime.h>

#define SEQ 2048
#define BATCH 2
#define NHEAD 16
#define HDIM 64
#define DMODEL 1024
#define N_QKV 3072
#define ROWS (BATCH * SEQ) // 4096

typedef __attribute__((ext_vector_type(16))) __bf16 v16bf;
typedef __attribute__((ext_vector_type(8)))  float  v8f;
typedef __attribute__((ext_vector_type(16))) unsigned short v16u;
typedef __attribute__((ext_vector_type(8)))  unsigned short v8u;
typedef __attribute__((ext_vector_type(4)))  unsigned short v4u;

union FragU { v16u u; v16bf b; };

__device__ inline unsigned short f2bf(float f) {
  unsigned int u = __float_as_uint(f);
  u += 0x7fffu + ((u >> 16) & 1u);           // round-to-nearest-even
  return (unsigned short)(u >> 16);
}

__device__ inline v8f wmma_bf16(v16bf a, v16bf b, v8f c) {
  // 8 args: (neg_a, A, neg_b, B, c_mod, C, reuse_a, reuse_b)
  return __builtin_amdgcn_wmma_f32_16x16x32_bf16(false, a, false, b, (short)0, c,
                                                 false, false);
}

// CDNA5 async global->LDS copy (ASYNCcnt-tracked, no VGPR data path).
__device__ inline void async_ld_lds_b128(unsigned int ldsoff, const void* gptr) {
  asm volatile("global_load_async_to_lds_b128 %0, %1, off"
               :: "v"(ldsoff), "v"(gptr) : "memory");
}
__device__ inline void wait_async0() {
  asm volatile("s_wait_asynccnt 0" ::: "memory");
}

// A fragment (16x32 bf16). ISA layout: lane m=l&15 holds row m; lanes 0-15 carry
// K chunks [0..7],[16..23]; lanes 16-31 carry [8..15],[24..31]. src row-major.
__device__ inline v16bf load_afrag(const unsigned short* src, int ld, int m0,
                                   int k0, int lane) {
  int m = lane & 15, hi = lane >> 4;
  const unsigned short* p = src + (size_t)(m0 + m) * ld + k0 + hi * 8;
  v8u a = *(const v8u*)p;
  v8u b = *(const v8u*)(p + 16);
  FragU f;
#pragma unroll
  for (int i = 0; i < 8; ++i) { f.u[i] = a[i]; f.u[i + 8] = b[i]; }
  return f.b;
}

// B fragment (32x16 bf16), logical B[k][n] = src[(n0+n)*ld + kk0 + k]
// (two 16B loads per lane). Lanes 0-15 carry K [0..7],[16..23]; 16-31 [8..15],[24..31].
__device__ inline v16bf load_bfrag_kc(const unsigned short* src, int ld, int n0,
                                      int kk0, int lane) {
  int n = lane & 15, hi = lane >> 4;
  const unsigned short* p = src + (size_t)(n0 + n) * ld + kk0 + hi * 8;
  v8u a = *(const v8u*)p;
  v8u b = *(const v8u*)(p + 16);
  FragU f;
#pragma unroll
  for (int i = 0; i < 8; ++i) { f.u[i] = a[i]; f.u[i + 8] = b[i]; }
  return f.b;
}

// ------------- Kernel 0a: x f32 -> bf16 (one pass, vectorized) --------------
__global__ void __launch_bounds__(256) cvt_x_kernel(
    const float* __restrict__ in, unsigned short* __restrict__ outb) {
  int i = (blockIdx.x * 256 + threadIdx.x) * 4;
  float4 v = *(const float4*)(in + i);
  v4u o;
  o[0] = f2bf(v.x); o[1] = f2bf(v.y); o[2] = f2bf(v.z); o[3] = f2bf(v.w);
  *(v4u*)(outb + i) = o;
}

// ------------- Kernel 0b: W[K,N] f32 -> WT[N,K] bf16 (tiled transpose) ------
__global__ void __launch_bounds__(256) cvt_transpose_kernel(
    const float* __restrict__ W, unsigned short* __restrict__ WT, int K, int N) {
  __shared__ unsigned short tile[32][33];
  int n0 = blockIdx.x * 32, k0 = blockIdx.y * 32;
  int tn = threadIdx.x & 31, tk = threadIdx.x >> 5;   // tk = 0..7
#pragma unroll
  for (int i = 0; i < 4; ++i) {
    int k = tk + i * 8;
    tile[k][tn] = f2bf(W[(size_t)(k0 + k) * N + n0 + tn]);
  }
  __syncthreads();
#pragma unroll
  for (int i = 0; i < 4; ++i) {
    int nn = tk + i * 8;
    WT[(size_t)(n0 + nn) * K + k0 + tn] = tile[tn][nn];   // coalesced in K
  }
}

// ------------- Kernel 1: QKV GEMM (bf16 x, K-major bf16 WT) -----------------
__global__ void __launch_bounds__(256) qkv_gemm_kernel(
    const unsigned short* __restrict__ xb, const unsigned short* __restrict__ WT,
    const float* __restrict__ bqkv, unsigned short* __restrict__ Qb,
    unsigned short* __restrict__ Kb, unsigned short* __restrict__ Vt) {
  __shared__ unsigned short As[128 * 40];  // 128 rows x 32 k (padded)
  int tid = threadIdx.x;
  int w = tid >> 5, lane = tid & 31;
  int lo = lane & 15, hi = lane >> 4;
  int n0 = blockIdx.x * 64, m0 = blockIdx.y * 128;
  int arow = tid >> 2, asub = (tid & 3) * 8;   // 16B chunk per async
  unsigned int lds0 = (unsigned int)(unsigned long long)&As[arow * 40 + asub];
  unsigned int lds1 =
      (unsigned int)(unsigned long long)&As[(arow + 64) * 40 + asub];

  v8f acc[4];
#pragma unroll
  for (int t = 0; t < 4; ++t)
#pragma unroll
    for (int i = 0; i < 8; ++i) acc[t][i] = 0.0f;

  for (int ks = 0; ks < DMODEL / 32; ++ks) {
    int k0 = ks * 32;
    const unsigned short* g0 = xb + (size_t)(m0 + arow) * DMODEL + k0 + asub;
    async_ld_lds_b128(lds0, g0);
    async_ld_lds_b128(lds1, g0 + (size_t)64 * DMODEL);
    wait_async0();
    __syncthreads();
    v16bf af = load_afrag(As, 40, w * 16, 0, lane);
#pragma unroll
    for (int t = 0; t < 4; ++t) {
      v16bf bf = load_bfrag_kc(WT, DMODEL, n0 + t * 16, k0, lane);  // from L2
      acc[t] = wmma_bf16(af, bf, acc[t]);
    }
    __syncthreads();
  }

  // epilogue: bias, scatter to Q[B,H,T,64], K[B,H,T,64], Vt[B,H,64,T]
#pragma unroll
  for (int t = 0; t < 4; ++t) {
    int n = n0 + t * 16 + lo;
    float bias = bqkv[n];
    int which = n >> 10, hh = (n >> 6) & 15, d = n & 63;
#pragma unroll
    for (int r = 0; r < 8; ++r) {
      int m = m0 + w * 16 + r + hi * 8;
      int b = m >> 11, tt = m & (SEQ - 1);
      int bh = b * NHEAD + hh;
      unsigned short val = f2bf(acc[t][r] + bias);
      if (which == 0)      Qb[((size_t)bh * SEQ + tt) * HDIM + d] = val;
      else if (which == 1) Kb[((size_t)bh * SEQ + tt) * HDIM + d] = val;
      else                 Vt[((size_t)bh * HDIM + d) * SEQ + tt] = val;
    }
  }
}

// ------------- Kernel 2: causal flash attention (one wave per q-tile) -------
__global__ void __launch_bounds__(256) attn_kernel(
    const unsigned short* __restrict__ Qb, const unsigned short* __restrict__ Kb,
    const unsigned short* __restrict__ Vt, unsigned short* __restrict__ Ab) {
  __shared__ unsigned short Plds[8][16 * 40];  // per-wave P staging tile
  int tid = threadIdx.x;
  int w = tid >> 5, lane = tid & 31;
  int lo = lane & 15, hi = lane >> 4;
  int gw = blockIdx.x * 8 + w;
  int qt = gw & 127;
  int bh = gw >> 7;
  int b = bh >> 4, h = bh & 15;
  const unsigned short* Q = Qb + (size_t)bh * SEQ * HDIM;
  const unsigned short* K = Kb + (size_t)bh * SEQ * HDIM;
  const unsigned short* V = Vt + (size_t)bh * HDIM * SEQ;
  unsigned short* myP = &Plds[w][0];
  int row0 = qt * 16;

  v16bf qf0 = load_afrag(Q, HDIM, row0, 0, lane);
  v16bf qf1 = load_afrag(Q, HDIM, row0, 32, lane);

  float mrow[8], srow[8], p0[8], p1[8], alpha[8];
  v8f acc[4];
#pragma unroll
  for (int r = 0; r < 8; ++r) { mrow[r] = -3.0e38f; srow[r] = 0.0f; }
#pragma unroll
  for (int t = 0; t < 4; ++t)
#pragma unroll
    for (int i = 0; i < 8; ++i) acc[t][i] = 0.0f;

  const float scale = 0.125f;  // 1/sqrt(64)
  int kpmax = qt >> 1;
  for (int kp = 0; kp <= kpmax; ++kp) {
    int k0 = kp * 32;
    v8f S0, S1;
#pragma unroll
    for (int i = 0; i < 8; ++i) { S0[i] = 0.0f; S1[i] = 0.0f; }
#pragma unroll
    for (int dstep = 0; dstep < 2; ++dstep) {
      v16bf a = dstep ? qf1 : qf0;
      v16bf b0 = load_bfrag_kc(K, HDIM, k0, dstep * 32, lane);
      v16bf b1 = load_bfrag_kc(K, HDIM, k0 + 16, dstep * 32, lane);
      S0 = wmma_bf16(a, b0, S0);
      S1 = wmma_bf16(a, b1, S1);
    }
    int col0 = k0 + lo, col1 = col0 + 16;
#pragma unroll
    for (int r = 0; r < 8; ++r) {
      int q = row0 + r + hi * 8;
      float s0 = S0[r] * scale, s1 = S1[r] * scale;
      if (col0 > q) s0 = -3.0e38f;      // causal mask
      if (col1 > q) s1 = -3.0e38f;
      float mx = fmaxf(s0, s1);
      mx = fmaxf(mx, __shfl_xor(mx, 1, 32));
      mx = fmaxf(mx, __shfl_xor(mx, 2, 32));
      mx = fmaxf(mx, __shfl_xor(mx, 4, 32));
      mx = fmaxf(mx, __shfl_xor(mx, 8, 32));
      float mnew = fmaxf(mrow[r], mx);
      float al = __expf(mrow[r] - mnew);
      float e0 = __expf(s0 - mnew);
      float e1 = __expf(s1 - mnew);
      float rs = e0 + e1;
      rs += __shfl_xor(rs, 1, 32);
      rs += __shfl_xor(rs, 2, 32);
      rs += __shfl_xor(rs, 4, 32);
      rs += __shfl_xor(rs, 8, 32);
      srow[r] = srow[r] * al + rs;
      mrow[r] = mnew;
      alpha[r] = al; p0[r] = e0; p1[r] = e1;
    }
#pragma unroll
    for (int t = 0; t < 4; ++t)
#pragma unroll
      for (int r = 0; r < 8; ++r) acc[t][r] *= alpha[r];
    // C-layout -> A-layout via per-wave LDS round trip
#pragma unroll
    for (int r = 0; r < 8; ++r) {
      myP[(r + hi * 8) * 40 + lo] = f2bf(p0[r]);
      myP[(r + hi * 8) * 40 + 16 + lo] = f2bf(p1[r]);
    }
    asm volatile("s_wait_dscnt 0" ::: "memory");
    v16bf pf = load_afrag(myP, 40, 0, 0, lane);
#pragma unroll
    for (int t = 0; t < 4; ++t) {
      v16bf vf = load_bfrag_kc(V, SEQ, t * 16, k0, lane);  // Vt is K-contiguous
      acc[t] = wmma_bf16(pf, vf, acc[t]);
    }
    asm volatile("" ::: "memory");
  }

  int tglob = b * SEQ + row0;
#pragma unroll
  for (int t = 0; t < 4; ++t) {
    int d = h * HDIM + t * 16 + lo;
#pragma unroll
    for (int r = 0; r < 8; ++r) {
      float o = acc[t][r] / srow[r];
      Ab[(size_t)(tglob + r + hi * 8) * DMODEL + d] = f2bf(o);
    }
  }
}

// ------------- Kernel 3: output projection GEMM (bf16 in, f32 out) ----------
__global__ void __launch_bounds__(256) out_gemm_kernel(
    const unsigned short* __restrict__ Ab, const unsigned short* __restrict__ WT,
    const float* __restrict__ bout, float* __restrict__ out) {
  __shared__ unsigned short As[128 * 40];
  int tid = threadIdx.x;
  int w = tid >> 5, lane = tid & 31;
  int lo = lane & 15, hi = lane >> 4;
  int n0 = blockIdx.x * 64, m0 = blockIdx.y * 128;
  int arow = tid >> 2, asub = (tid & 3) * 8;
  unsigned int lds0 = (unsigned int)(unsigned long long)&As[arow * 40 + asub];
  unsigned int lds1 =
      (unsigned int)(unsigned long long)&As[(arow + 64) * 40 + asub];

  v8f acc[4];
#pragma unroll
  for (int t = 0; t < 4; ++t)
#pragma unroll
    for (int i = 0; i < 8; ++i) acc[t][i] = 0.0f;

  for (int ks = 0; ks < DMODEL / 32; ++ks) {
    int k0 = ks * 32;
    const unsigned short* g0 = Ab + (size_t)(m0 + arow) * DMODEL + k0 + asub;
    async_ld_lds_b128(lds0, g0);
    async_ld_lds_b128(lds1, g0 + (size_t)64 * DMODEL);
    wait_async0();
    __syncthreads();
    v16bf af = load_afrag(As, 40, w * 16, 0, lane);
#pragma unroll
    for (int t = 0; t < 4; ++t) {
      v16bf bf = load_bfrag_kc(WT, DMODEL, n0 + t * 16, k0, lane);
      acc[t] = wmma_bf16(af, bf, acc[t]);
    }
    __syncthreads();
  }

#pragma unroll
  for (int t = 0; t < 4; ++t) {
    int n = n0 + t * 16 + lo;
    float bias = bout[n];
#pragma unroll
    for (int r = 0; r < 8; ++r) {
      int m = m0 + w * 16 + r + hi * 8;
      out[(size_t)m * DMODEL + n] = acc[t][r] + bias;
    }
  }
}

extern "C" void kernel_launch(void* const* d_in, const int* in_sizes, int n_in,
                              void* d_out, int out_size, void* d_ws,
                              size_t ws_size, hipStream_t stream) {
  const float* x    = (const float*)d_in[0];
  const float* Wqkv = (const float*)d_in[1];
  const float* bqkv = (const float*)d_in[2];
  const float* Wout = (const float*)d_in[3];
  const float* bout = (const float*)d_in[4];
  float* out = (float*)d_out;

  unsigned short* ws = (unsigned short*)d_ws;
  const size_t QKV_ELEMS = (size_t)BATCH * NHEAD * SEQ * HDIM;  // 4,194,304
  unsigned short* Qb    = ws;
  unsigned short* Kb    = Qb + QKV_ELEMS;
  unsigned short* Vt    = Kb + QKV_ELEMS;
  unsigned short* Ab    = Vt + QKV_ELEMS;
  unsigned short* xb    = Ab + QKV_ELEMS;                 // [4096][1024]
  unsigned short* WqkvT = xb + (size_t)ROWS * DMODEL;     // [3072][1024]
  unsigned short* WoutT = WqkvT + (size_t)N_QKV * DMODEL; // [1024][1024]

  cvt_x_kernel<<<dim3((ROWS * DMODEL) / (256 * 4)), 256, 0, stream>>>(x, xb);
  cvt_transpose_kernel<<<dim3(N_QKV / 32, DMODEL / 32), 256, 0, stream>>>(
      Wqkv, WqkvT, DMODEL, N_QKV);
  cvt_transpose_kernel<<<dim3(DMODEL / 32, DMODEL / 32), 256, 0, stream>>>(
      Wout, WoutT, DMODEL, DMODEL);
  qkv_gemm_kernel<<<dim3(N_QKV / 64, ROWS / 128), 256, 0, stream>>>(
      xb, WqkvT, bqkv, Qb, Kb, Vt);
  attn_kernel<<<dim3((BATCH * NHEAD * (SEQ / 16)) / 8), 256, 0, stream>>>(
      Qb, Kb, Vt, Ab);
  out_gemm_kernel<<<dim3(DMODEL / 64, ROWS / 128), 256, 0, stream>>>(
      Ab, WoutT, bout, out);
}
